// QNetNode_71554155152032
// MI455X (gfx1250) — matrix operand order repaired and checked by reading, not compiled, not run on hardware
//
#include <hip/hip_runtime.h>

// ---------------------------------------------------------------------------
// QNetNode forward for MI455X (gfx1250, wave32, WMMA bf16 -> f32 accumulate)
//
//   prep_weights : f32 weights -> bf16, pre-swizzled into WMMA B-frag order
//   base_gemm    : base = node_features @ w_n2l + bias_n2l          (WMMA)
//   per graph b (8x):
//     h0_kernel  : h = relu(base + onehot(picked)*bias_picked)
//                  (f32 row-major for SpMM  +  bf16 in A-frag order)
//     2x: memset pooled; spmm (gather + global f32 atomics);
//         conv_gemm: h = relu(pooled@conv_W + conv_b + input_msg)   (WMMA)
//     colsum / prep_final : graph constants  v, gbias, c
//     final_gemm : q[n] = sum_j relu((h@W1a)[n,j]+gbias[j])*v[j]+c
//                  bf16 A tile DMA'd by TDM tensor_load_to_lds (contiguous,
//                  already in fragment order), then WMMA + LDS reduce.
//
// All GEMM operands (A via LDS, B via registers) are stored in exact WMMA
// fragment order, so every fragment load is one contiguous 32B per-lane read
// (2x ds_load_b128 / global_load_b128) with zero repacking VALU.
// ---------------------------------------------------------------------------

typedef __bf16        v16bf __attribute__((ext_vector_type(16)));
typedef float         v8f   __attribute__((ext_vector_type(8)));
typedef unsigned int  u32x4 __attribute__((ext_vector_type(4)));
typedef unsigned int  u32x8 __attribute__((ext_vector_type(8)));
typedef int           i32x4 __attribute__((ext_vector_type(4)));
typedef int           i32x8 __attribute__((ext_vector_type(8)));

#define NNODES 20000
#define NEDGES 320000
#define DLAT   64
#define FDIM   128
#define HDIM   128
#define MROWS  80            // rows per block = 5 subtiles of 16 (20000 = 250*80)
#define MT     5

__device__ __forceinline__ unsigned short f2bf(float x) {
  unsigned int u = __builtin_bit_cast(unsigned int, x);
  u += 0x7FFFu + ((u >> 16) & 1u);   // round-to-nearest-even
  return (unsigned short)(u >> 16);
}

// Fragment-order addressing (16-bit A 16x32 WMMA layout):
//   lane = (row%16) + 16*((k%32)>>3 & 1 ... ) -- concretely:
//   within a 16x32 chunk, lane l holds rows l&15, K-half l>>4; vreg v holds
//   k = 16*(v>=4) + 8*(l>>4) + 2*(v&3) + {0,1}.
// Storage: dword[((mt*KC + kc)*32 + lane)*8 + v] = pack(elem(k), elem(k+1)).

// A fragment: one contiguous 32B per-lane read from fragment-ordered LDS.
__device__ __forceinline__ v16bf load_afrag_sw(const unsigned int* AS, int mtkc) {
  const int lane = threadIdx.x & 31;
  u32x8 d = *(const u32x8*)(AS + (mtkc * 32 + lane) * 8);
  return __builtin_bit_cast(v16bf, d);
}

// B fragments from pre-swizzled weights: two global_load_b128 per chunk.
template <int K>
__device__ __forceinline__ void load_bfrags(const unsigned int* Bswz, int colTile,
                                            v16bf* b) {
  const int lane = threadIdx.x & 31;
  const unsigned int* p = Bswz + (size_t)colTile * (K / 32) * 256;
#pragma unroll
  for (int kc = 0; kc < K / 32; ++kc) {
    u32x8 d = *(const u32x8*)(p + kc * 256 + lane * 8);
    b[kc] = __builtin_bit_cast(v16bf, d);
  }
}

// Stage f32 tile (MROWS x K, row-major global) into fragment-ordered LDS dwords.
template <int K, int NTHREADS>
__device__ __forceinline__ void stage_A(const float* __restrict__ src, int row0,
                                        unsigned int* AS) {
  constexpr int KC = K / 32;
  for (int idx = threadIdx.x; idx < MROWS * K / 2; idx += NTHREADS) {
    const int v = idx & 7, lane = (idx >> 3) & 31, chunk = idx >> 8;
    const int kc = chunk % KC, mtl = chunk / KC;
    const int row = mtl * 16 + (lane & 15);
    const int ka = (kc << 5) + ((v & 4) << 2) + ((lane >> 4) << 3) + ((v & 3) << 1);
    const float* p = src + (size_t)(row0 + row) * K + ka;
    unsigned int lo = f2bf(p[0]), hi = f2bf(p[1]);
    AS[idx] = lo | (hi << 16);
  }
}

// Element offset of (row-in-block rem, feature d) inside an 80x64 bf16 block
// stored in fragment order (KC = 2 chunks of 32 along d).
__device__ __forceinline__ int hbf_local(int rem, int d) {
  const int mt = rem >> 4, rowl = rem & 15;
  const int kc = d >> 5, cl = d & 31;
  const int lane = rowl + (((cl >> 3) & 1) << 4);
  const int v = ((cl & 16) >> 2) | ((cl >> 1) & 3);
  const int dw = (((mt << 1) + kc) * 32 + lane) * 8 + v;
  return dw * 2 + (d & 1);
}

// ------------- weight conversion + swizzle (once per launch) ---------------
__device__ __forceinline__ void swizzleB(const float* __restrict__ W,
                                         unsigned int* __restrict__ out,
                                         int K, int C, int idx) {
  const int v = idx & 7;
  const int lane = (idx >> 3) & 31;
  const int chunk = idx >> 8;
  const int KC = K >> 5;
  const int kc = chunk % KC, ct = chunk / KC;
  const int kb = (kc << 5) + ((v & 4) << 2) + ((lane >> 4) << 3) + ((v & 3) << 1);
  const int col = ct * 16 + (lane & 15);
  unsigned int lo = f2bf(W[kb * C + col]);
  unsigned int hi = f2bf(W[(kb + 1) * C + col]);
  out[idx] = lo | (hi << 16);
}

__global__ void prep_weights(const float* __restrict__ w_n2l,
                             const float* __restrict__ conv_W,
                             const float* __restrict__ lin1_W,
                             unsigned int* __restrict__ w_n2l_sw,
                             unsigned int* __restrict__ conv_W_sw,
                             unsigned int* __restrict__ lin1a_sw) {
  int i = blockIdx.x * blockDim.x + threadIdx.x;
  if (i < FDIM * DLAT / 2) swizzleB(w_n2l, w_n2l_sw, FDIM, DLAT, i);
  if (i < DLAT * DLAT / 2) swizzleB(conv_W, conv_W_sw, DLAT, DLAT, i);
  if (i < DLAT * HDIM / 2) swizzleB(lin1_W, lin1a_sw, DLAT, HDIM, i);
}

// ---------------- base = nf @ w_n2l + bias_n2l  (WMMA, 80-row tile) --------
__global__ void __launch_bounds__(128)
base_gemm(const float* __restrict__ nf, const unsigned int* __restrict__ Wsw,
          const float* __restrict__ bias, float* __restrict__ base) {
  __shared__ __align__(32) unsigned int AS[MT * (FDIM / 32) * 256];  // 20KB
  const int row0 = blockIdx.x * MROWS;
  const int wave = threadIdx.x >> 5, lane = threadIdx.x & 31;
  const int hf = lane >> 4, rl = lane & 15;
  v16bf bfr[FDIM / 32];
  load_bfrags<FDIM>(Wsw, wave, bfr);           // hoisted across 5 subtiles
  stage_A<FDIM, 128>(nf, row0, AS);
  __syncthreads();
  const int col = wave * 16 + rl;
  const float bc = bias[col];
#pragma unroll
  for (int mt = 0; mt < MT; ++mt) {
    v8f acc = {0.f, 0.f, 0.f, 0.f, 0.f, 0.f, 0.f, 0.f};
#pragma unroll
    for (int kc = 0; kc < FDIM / 32; ++kc)
      acc = __builtin_amdgcn_wmma_f32_16x16x32_bf16(
          false, load_afrag_sw(AS, mt * (FDIM / 32) + kc), false, bfr[kc],
          (short)0, acc, false, false);
#pragma unroll
    for (int r = 0; r < 8; ++r) {
      int m = mt * 16 + r + (hf << 3);
      base[(size_t)(row0 + m) * DLAT + col] = acc[r] + bc;
    }
  }
}

// ------- h0 = relu(base + onehot(picked)*bias_picked); also bf16-swizzled --
__global__ void h0_kernel(const float* __restrict__ base,
                          const float* __restrict__ bias_picked,
                          const int* __restrict__ picked, int b,
                          float* __restrict__ h, unsigned short* __restrict__ h_bf) {
  int i = blockIdx.x * blockDim.x + threadIdx.x;
  if (i >= NNODES * DLAT) return;
  int n = i >> 6, d = i & 63;
  float x = base[i];
  if (n == picked[b]) x += bias_picked[d];
  x = fmaxf(x, 0.f);
  h[i] = x;
  int blk = n / MROWS, rem = n % MROWS;
  h_bf[(size_t)blk * (MROWS * DLAT) + hbf_local(rem, d)] = f2bf(x);
}

// ---------------- pooled[dst] += w * h[src]  (gather + f32 atomics) --------
__global__ void spmm(const float* __restrict__ h, const int* __restrict__ src,
                     const int* __restrict__ dst, const float* __restrict__ w,
                     float* __restrict__ pooled) {
  int t = blockIdx.x * blockDim.x + threadIdx.x;
  int e = t >> 4;            // 16 lanes per edge, 4 floats per lane
  int part = t & 15;
  if (e >= NEDGES) return;
  int s = src[e], d = dst[e];
  float wt = w[e];
  float4 hv = *(const float4*)(h + (size_t)s * DLAT + part * 4);
  float* pd = pooled + (size_t)d * DLAT + part * 4;
  atomicAdd(pd + 0, hv.x * wt);
  atomicAdd(pd + 1, hv.y * wt);
  atomicAdd(pd + 2, hv.z * wt);
  atomicAdd(pd + 3, hv.w * wt);
}

// ------- h = relu(pooled @ conv_W + conv_b + input_message)  (WMMA) --------
__global__ void __launch_bounds__(128)
conv_gemm(const float* __restrict__ pooled, const unsigned int* __restrict__ Wsw,
          const float* __restrict__ conv_b, const float* __restrict__ base,
          const float* __restrict__ bias_picked, const int* __restrict__ picked,
          int b, float* __restrict__ h, unsigned short* __restrict__ h_bf) {
  __shared__ __align__(32) unsigned int AS[MT * (DLAT / 32) * 256];   // 10KB
  const int row0 = blockIdx.x * MROWS;
  const int wave = threadIdx.x >> 5, lane = threadIdx.x & 31;
  const int hf = lane >> 4, rl = lane & 15;
  v16bf bfr[DLAT / 32];
  load_bfrags<DLAT>(Wsw, wave, bfr);
  stage_A<DLAT, 128>(pooled, row0, AS);
  __syncthreads();
  const int col = wave * 16 + rl;
  const float cb = conv_b[col];
  const int pk = picked[b];
  unsigned short* hb = h_bf + (size_t)blockIdx.x * (MROWS * DLAT);
#pragma unroll
  for (int mt = 0; mt < MT; ++mt) {
    v8f acc = {0.f, 0.f, 0.f, 0.f, 0.f, 0.f, 0.f, 0.f};
#pragma unroll
    for (int kc = 0; kc < DLAT / 32; ++kc)
      acc = __builtin_amdgcn_wmma_f32_16x16x32_bf16(
          false, load_afrag_sw(AS, mt * (DLAT / 32) + kc), false, bfr[kc],
          (short)0, acc, false, false);
#pragma unroll
    for (int r = 0; r < 8; ++r) {
      int rem = mt * 16 + r + (hf << 3);
      int n = row0 + rem;
      float x = acc[r] + cb + base[(size_t)n * DLAT + col];
      if (n == pk) x += bias_picked[col];
      x = fmaxf(x, 0.f);
      h[(size_t)n * DLAT + col] = x;
      hb[hbf_local(rem, col)] = f2bf(x);
    }
  }
}

// ---------------- column sums of h -> gsum[64] -----------------------------
__global__ void colsum(const float* __restrict__ h, float* __restrict__ gsum,
                       int rowsPerBlock) {
  __shared__ float acc[DLAT];
  int tid = threadIdx.x;
  if (tid < DLAT) acc[tid] = 0.f;
  __syncthreads();
  int col = tid & 63, rlane = tid >> 6;
  int r0 = blockIdx.x * rowsPerBlock;
  float s = 0.f;
  for (int r = r0 + rlane; r < r0 + rowsPerBlock; r += 4)
    s += h[(size_t)r * DLAT + col];
  atomicAdd(&acc[col], s);
  __syncthreads();
  if (tid < DLAT) atomicAdd(&gsum[tid], acc[tid]);
}

// ----- per-graph readout constants: v[128], gbias[128], c ------------------
__global__ void prep_final(const float* __restrict__ h, const float* __restrict__ gsum,
                           const int* __restrict__ tgt_nodes, int b,
                           const float* __restrict__ lin1_W, const float* __restrict__ lin1_b,
                           const float* __restrict__ lout_W, const float* __restrict__ lout_b,
                           float* __restrict__ v, float* __restrict__ gbias,
                           float* __restrict__ cconst) {
  __shared__ float t[DLAT];
  __shared__ float g[DLAT];
  int j = threadIdx.x;  // 128 threads
  int tgt = tgt_nodes[b];
  if (j < DLAT) {
    t[j] = h[(size_t)tgt * DLAT + j];
    g[j] = gsum[j] * (1.0f / (float)NNODES);
  }
  __syncthreads();
  float vj = 0.f, gb = lin1_b[j];
  for (int d = 0; d < DLAT; ++d) {
    vj += lout_W[j * DLAT + d] * t[d];                 // lout_W[H,D]
    gb += g[d] * lin1_W[(DLAT + d) * HDIM + j];        // lin1_W rows 64..127
  }
  v[j] = vj;
  gbias[j] = gb;
  if (j == 0) {
    float cc = 0.f;
    for (int d = 0; d < DLAT; ++d) cc += lout_b[d] * t[d];
    *cconst = cc;
  }
}

// -- q[n] = sum_j relu((h@W1a)[n,j]+gbias[j])*v[j] + c ----------------------
// bf16 A tile is already in fragment order in memory; TDM DMAs the whole
// 10240B block into LDS (1-D tile, 2B elements), then fragments are plain
// contiguous 32B per-lane reads.
__global__ void __launch_bounds__(256)
final_gemm(const unsigned short* __restrict__ h_bf,
           const unsigned int* __restrict__ W1sw,
           const float* __restrict__ gbias, const float* __restrict__ v,
           const float* __restrict__ cconst, float* __restrict__ q) {
  __shared__ __align__(32) unsigned int AS[MT * (DLAT / 32) * 256];   // 10KB
  __shared__ float qacc[MROWS];
  const int row0 = blockIdx.x * MROWS;
  const int wave = threadIdx.x >> 5, lane = threadIdx.x & 31;
  const int hf = lane >> 4, rl = lane & 15;

  if (threadIdx.x < 32) {  // wave-uniform guard: only wave 0 issues the DMA
    unsigned long long ga =
        (unsigned long long)(const void*)(h_bf + (size_t)blockIdx.x * (MROWS * DLAT));
    unsigned int ldsoff = (unsigned int)(unsigned long long)(void*)AS;
    u32x4 g0 = {1u,                                    // count=1 valid D#
                ldsoff,                                // lds_addr
                (unsigned int)ga,                      // global_addr[31:0]
                (unsigned int)((ga >> 32) & 0x01FFFFFFu) | (2u << 30)};  // type=2
    // g1: data_size=2B (1<<16); tensor_dim0=5120, tensor_dim1=1,
    //     tile_dim0=5120 (1-D contiguous), dim0_stride=5120.
    i32x8 g1 = {(int)(1u << 16),
                (int)(5120u << 16),        // tensor_dim0[15:0] @ bits 63:48
                (int)(1u << 16),           // dim0[31:16]=0, tensor_dim1[15:0]=1
                (int)(5120u << 16),        // dim1[31:16]=0, tile_dim0=5120
                0,                         // tile_dim1=0, tile_dim2=0
                5120,                      // tensor_dim0_stride[31:0]
                0, 0};
    i32x4 z4 = {0, 0, 0, 0};
#if __clang_major__ >= 23
    i32x8 z8 = {0, 0, 0, 0, 0, 0, 0, 0};
    __builtin_amdgcn_tensor_load_to_lds(g0, g1, z4, z4, z8, 0);
#else
    __builtin_amdgcn_tensor_load_to_lds(g0, g1, z4, z4, 0);
#endif
    __builtin_amdgcn_s_wait_tensorcnt(0);
  }
  v16bf bfr[DLAT / 32];
  load_bfrags<DLAT>(W1sw, wave, bfr);      // overlaps with the TDM transfer
  if (threadIdx.x < MROWS) qacc[threadIdx.x] = 0.f;
  __syncthreads();                          // TDM data + qacc init visible

  const int col = wave * 16 + rl;
  const float vc = v[col], gb = gbias[col];
#pragma unroll
  for (int mt = 0; mt < MT; ++mt) {
    v8f acc = {0.f, 0.f, 0.f, 0.f, 0.f, 0.f, 0.f, 0.f};
#pragma unroll
    for (int kc = 0; kc < DLAT / 32; ++kc)
      acc = __builtin_amdgcn_wmma_f32_16x16x32_bf16(
          false, load_afrag_sw(AS, mt * (DLAT / 32) + kc), false, bfr[kc],
          (short)0, acc, false, false);
#pragma unroll
    for (int r = 0; r < 8; ++r) {
      float x = fmaxf(acc[r] + gb, 0.f);
      atomicAdd(&qacc[mt * 16 + r + (hf << 3)], x * vc);   // LDS ds_add_f32
    }
  }
  __syncthreads();
  if (threadIdx.x < MROWS)
    q[row0 + threadIdx.x] = qacc[threadIdx.x] + *cconst;
}

// ---------------------------------------------------------------------------
extern "C" void kernel_launch(void* const* d_in, const int* in_sizes, int n_in,
                              void* d_out, int out_size, void* d_ws, size_t ws_size,
                              hipStream_t stream) {
  (void)in_sizes; (void)n_in; (void)out_size; (void)ws_size;
  const float* nf          = (const float*)d_in[0];
  const float* w_n2l       = (const float*)d_in[1];
  const float* bias_n2l    = (const float*)d_in[2];
  const float* bias_picked = (const float*)d_in[3];
  const float* conv_W      = (const float*)d_in[4];
  const float* conv_b      = (const float*)d_in[5];
  const float* lin1_W      = (const float*)d_in[6];
  const float* lin1_b      = (const float*)d_in[7];
  const float* lout_W      = (const float*)d_in[8];
  const float* lout_b      = (const float*)d_in[9];
  const float* edge_w      = (const float*)d_in[10];
  const int*   esrc        = (const int*)d_in[11];
  const int*   edst        = (const int*)d_in[12];
  const int*   tgt         = (const int*)d_in[13];
  const int*   picked      = (const int*)d_in[14];
  float*       out         = (float*)d_out;

  char* ws = (char*)d_ws;
  float* base   = (float*)ws;  ws += (size_t)NNODES * DLAT * 4;
  float* h      = (float*)ws;  ws += (size_t)NNODES * DLAT * 4;
  float* pooled = (float*)ws;  ws += (size_t)NNODES * DLAT * 4;
  float* gsum   = (float*)ws;  ws += DLAT * 4;
  float* vvec   = (float*)ws;  ws += HDIM * 4;
  float* gbias  = (float*)ws;  ws += HDIM * 4;
  float* cconst = (float*)ws;  ws += 32;                      // keep 32B align
  unsigned int* w_n2l_sw  = (unsigned int*)ws; ws += FDIM * DLAT * 2;
  unsigned int* conv_W_sw = (unsigned int*)ws; ws += DLAT * DLAT * 2;
  unsigned int* lin1a_sw  = (unsigned int*)ws; ws += DLAT * HDIM * 2;
  unsigned short* h_bf    = (unsigned short*)ws; ws += (size_t)NNODES * DLAT * 2;

  prep_weights<<<(FDIM * DLAT / 2 + 255) / 256, 256, 0, stream>>>(
      w_n2l, conv_W, lin1_W, w_n2l_sw, conv_W_sw, lin1a_sw);

  base_gemm<<<NNODES / MROWS, 128, 0, stream>>>(nf, w_n2l_sw, bias_n2l, base);

  for (int b = 0; b < 8; ++b) {
    h0_kernel<<<(NNODES * DLAT) / 256, 256, 0, stream>>>(base, bias_picked,
                                                         picked, b, h, h_bf);
    for (int lv = 0; lv < 2; ++lv) {
      hipMemsetAsync(pooled, 0, (size_t)NNODES * DLAT * 4, stream);
      spmm<<<(NEDGES * 16) / 256, 256, 0, stream>>>(
          h, esrc + (size_t)b * NEDGES, edst + (size_t)b * NEDGES,
          edge_w + (size_t)b * NEDGES, pooled);
      conv_gemm<<<NNODES / MROWS, 128, 0, stream>>>(
          pooled, conv_W_sw, conv_b, base, bias_picked, picked, b, h, h_bf);
    }
    hipMemsetAsync(gsum, 0, DLAT * 4, stream);
    colsum<<<100, 256, 0, stream>>>(h, gsum, NNODES / 100);
    prep_final<<<1, 128, 0, stream>>>(h, gsum, tgt, b, lin1_W, lin1_b,
                                      lout_W, lout_b, vvec, gbias, cconst);
    final_gemm<<<NNODES / MROWS, 256, 0, stream>>>(h_bf, lin1a_sw, gbias, vvec,
                                                   cconst, out + (size_t)b * NNODES);
  }
}
